// GraphActorCritic_74775380624040
// MI455X (gfx1250) — matrix operand order repaired and checked by reading, not compiled, not run on hardware
//
#include <hip/hip_runtime.h>

typedef __attribute__((ext_vector_type(16))) _Float16 v16h;
typedef __attribute__((ext_vector_type(2)))  __fp16   v2fp16;
typedef __attribute__((ext_vector_type(8)))  float    v8f;
typedef __attribute__((ext_vector_type(4)))  unsigned int u32x4;

// ---------------- workspace / LDS layout (bytes) ----------------
#define OFF_W1T   0        // [128][32]  f16 (K padded to 32)        8192 B
#define OFF_W2T   8192     // [128][128] f16                        32768 B
#define OFF_W3T   40960    // [128][128] f16                        32768 B
#define OFF_B1    73728    // [128] f32                               512 B
#define OFF_B2    74240
#define OFF_B3    74752
#define OFF_W4    75264    // [128] f16 packed (256B used of 512B)
#define OFF_B4    75776    // [1] f32
#define PARAM_BYTES 75780
#define PARAM_WORDS (PARAM_BYTES/4)     // 18945
#define MLP_STRIDE  76032               // 256-aligned stride in workspace

#define WS_MSG      0                    // 3200 f32 message accumulator
#define WS_PARAMS   12800                // 3 param blocks follow

#define OFF_HBUF    75792                // LDS: per-wave activation buffers
#define NWAVES      8
// msg kernel: two 16x128 f16 buffers per wave; actor/critic: one
#define LDS_BYTES_1T (OFF_HBUF + NWAVES*4096)   // 108560
#define LDS_BYTES_2T (OFF_HBUF + NWAVES*8192)   // 141328

#define WMMA_F16(A, B, C) \
    __builtin_amdgcn_wmma_f32_16x16x32_f16(false, (A), false, (B), (short)0, (C), false, false)

// ---------------- fragment loaders (LDS, contiguous k = 16*(lane/16) + s order) ------
__device__ __forceinline__ v16h ld_frag16(const _Float16* p0, const _Float16* p1) {
    union { u32x4 q[2]; v16h h; } u;
    u.q[0] = *(const u32x4*)p0;
    u.q[1] = *(const u32x4*)p1;
    return u.h;
}

// B fragment (32x16 f16) from Wt[n][k] (k contiguous, rowstride halves).
__device__ __forceinline__ v16h ld_b(const _Float16* w, int rowstride, int k0,
                                     int nt, int lane) {
    int n  = nt * 16 + (lane & 15);
    int ks = k0 + ((lane & 16) ? 16 : 0);
    const _Float16* base = w + n * rowstride + ks;
    return ld_frag16(base, base + 8);
}

// A fragment (16x32 f16) from h[m][k] row-major (row stride 128 halves).
__device__ __forceinline__ v16h ld_a(const _Float16* h, int kt, int lane) {
    int m  = lane & 15;
    int k0 = kt * 32 + ((lane & 16) ? 16 : 0);
    const _Float16* base = h + m * 128 + k0;
    return ld_frag16(base, base + 8);
}

// bias add + relu + packed f32->f16 conversion + store into h[m][n]
__device__ __forceinline__ void bias_relu_store(v8f c, float bv, _Float16* h,
                                                int nt, int lane) {
    int n  = nt * 16 + (lane & 15);
    int mb = (lane >> 4) * 8;
    unsigned short* hs = (unsigned short*)h;
#pragma unroll
    for (int r = 0; r < 8; r += 2) {
        union { v2fp16 v; unsigned short us[2]; } p;
        p.v = __builtin_amdgcn_cvt_pkrtz(fmaxf(c[r]     + bv, 0.0f),
                                         fmaxf(c[r + 1] + bv, 0.0f));
        hs[(mb + r    ) * 128 + n] = p.us[0];
        hs[(mb + r + 1) * 128 + n] = p.us[1];
    }
}

// Per-lane loop-invariant constants hoisted out of the tile loop.
struct TileConsts {
    float b1v[8], b2v[8], b3v[8];   // bias value for this lane's column, per ntile
    unsigned int w4p[32];           // layer-4 weights, f16x2, this lane's K-half
    float b4;
};

__device__ __forceinline__ void load_consts(TileConsts& tc, const char* smem,
                                            int lane) {
    const float* b1 = (const float*)(smem + OFF_B1);
    const float* b2 = (const float*)(smem + OFF_B2);
    const float* b3 = (const float*)(smem + OFF_B3);
    const unsigned int* w4p = (const unsigned int*)(smem + OFF_W4);
    int col = lane & 15;
#pragma unroll
    for (int nt = 0; nt < 8; ++nt) {
        tc.b1v[nt] = b1[nt * 16 + col];
        tc.b2v[nt] = b2[nt * 16 + col];
        tc.b3v[nt] = b3[nt * 16 + col];
    }
    int kh2 = (lane >> 4) * 32;     // dword offset of this lane's K-half
#pragma unroll
    for (int t = 0; t < 32; ++t) tc.w4p[t] = w4p[kh2 + t];
    tc.b4 = *(const float*)(smem + OFF_B4);
}

// layer-4 dot for one tile buffer: [16x128] @ [128x1]
__device__ __forceinline__ float dot_out(const _Float16* hbuf,
                                         const TileConsts& tc, int lane) {
    int m  = lane & 15;
    int kh = (lane >> 4) * 64;
    const unsigned int* hp = (const unsigned int*)(hbuf + m * 128 + kh);
    float s = 0.0f;
#pragma unroll
    for (int t = 0; t < 32; ++t) {
        union { unsigned int u; __fp16 h[2]; } hv, wv;
        hv.u = hp[t];
        wv.u = tc.w4p[t];
        s = fmaf((float)hv.h[0], (float)wv.h[0], s);
        s = fmaf((float)hv.h[1], (float)wv.h[1], s);
    }
    s += __shfl_xor(s, 16, 32);
    return s + tc.b4;
}

// ======================= single-tile path (actor / critic) ==================
__device__ __forceinline__ void dense128(const _Float16* wt, const float* bv,
                                         v8f czero, _Float16* hbuf, int lane) {
    v16h af[4];
#pragma unroll
    for (int kt = 0; kt < 4; ++kt) af[kt] = ld_a(hbuf, kt, lane);
    v8f acc[8];
#pragma unroll
    for (int nt = 0; nt < 8; ++nt) {
        acc[nt] = czero;
#pragma unroll
        for (int kt = 0; kt < 4; ++kt) {
            v16h bf = ld_b(wt, 128, kt * 32, nt, lane);
            acc[nt] = WMMA_F16(af[kt], bf, acc[nt]);
        }
    }
#pragma unroll
    for (int nt = 0; nt < 8; ++nt) bias_relu_store(acc[nt], bv[nt], hbuf, nt, lane);
}

__device__ __forceinline__ float mlp_tile(v16h a1,
        const _Float16* w1t, const _Float16* w2t, const _Float16* w3t,
        const TileConsts& tc, _Float16* hbuf, int lane) {
    v8f czero;
#pragma unroll
    for (int r = 0; r < 8; ++r) czero[r] = 0.0f;
    {
        v8f acc[8];
#pragma unroll
        for (int nt = 0; nt < 8; ++nt) {
            v16h bf = ld_b(w1t, 32, 0, nt, lane);
            acc[nt] = WMMA_F16(a1, bf, czero);
        }
#pragma unroll
        for (int nt = 0; nt < 8; ++nt)
            bias_relu_store(acc[nt], tc.b1v[nt], hbuf, nt, lane);
    }
    dense128(w2t, tc.b2v, czero, hbuf, lane);
    dense128(w3t, tc.b3v, czero, hbuf, lane);
    return dot_out(hbuf, tc, lane);
}

// ======================= dual-tile path (message kernel) ====================
// Two 16-row tiles share every staged B fragment: halves LDS B-traffic/WMMA
// and gives two independent accumulator chains per ntile.
__device__ __forceinline__ void dense128x2(const _Float16* wt, const float* bv,
                                           v8f czero, _Float16* hA, _Float16* hB,
                                           int lane) {
    v16h afA[4], afB[4];
#pragma unroll
    for (int kt = 0; kt < 4; ++kt) {
        afA[kt] = ld_a(hA, kt, lane);
        afB[kt] = ld_a(hB, kt, lane);
    }
#pragma unroll
    for (int nt = 0; nt < 8; ++nt) {
        v16h bf[4];
#pragma unroll
        for (int kt = 0; kt < 4; ++kt) bf[kt] = ld_b(wt, 128, kt * 32, nt, lane);
        v8f accA = czero, accB = czero;
#pragma unroll
        for (int kt = 0; kt < 4; ++kt) accA = WMMA_F16(afA[kt], bf[kt], accA);
#pragma unroll
        for (int kt = 0; kt < 4; ++kt) accB = WMMA_F16(afB[kt], bf[kt], accB);
        bias_relu_store(accA, bv[nt], hA, nt, lane);
        bias_relu_store(accB, bv[nt], hB, nt, lane);
    }
}

__device__ __forceinline__ float2 mlp_tile2(v16h a1a, v16h a1b,
        const _Float16* w1t, const _Float16* w2t, const _Float16* w3t,
        const TileConsts& tc, _Float16* hA, _Float16* hB, int lane) {
    v8f czero;
#pragma unroll
    for (int r = 0; r < 8; ++r) czero[r] = 0.0f;
    // layer 1: each B fragment feeds both tiles
#pragma unroll
    for (int nt = 0; nt < 8; ++nt) {
        v16h bf = ld_b(w1t, 32, 0, nt, lane);
        v8f accA = WMMA_F16(a1a, bf, czero);
        v8f accB = WMMA_F16(a1b, bf, czero);
        bias_relu_store(accA, tc.b1v[nt], hA, nt, lane);
        bias_relu_store(accB, tc.b1v[nt], hB, nt, lane);
    }
    dense128x2(w2t, tc.b2v, czero, hA, hB, lane);
    dense128x2(w3t, tc.b3v, czero, hA, hB, lane);
    float2 r;
    r.x = dot_out(hA, tc, lane);
    r.y = dot_out(hB, tc, lane);
    return r;
}

// Copy one param block from global workspace into LDS.
__device__ __forceinline__ void stage_params(const char* g, char* smem) {
    const unsigned int* src = (const unsigned int*)g;
    unsigned int* dst = (unsigned int*)smem;
    for (int i = threadIdx.x; i < PARAM_WORDS; i += blockDim.x) dst[i] = src[i];
    __syncthreads();
}

#define GET_PTRS()                                                     \
    const _Float16* w1t = (const _Float16*)(smem + OFF_W1T);           \
    const _Float16* w2t = (const _Float16*)(smem + OFF_W2T);           \
    const _Float16* w3t = (const _Float16*)(smem + OFF_W3T);           \
    int wave = threadIdx.x >> 5;                                       \
    int lane = threadIdx.x & 31;                                       \
    TileConsts tc;                                                     \
    load_consts(tc, smem, lane);

// ---------------- weight preparation: f32 row-major -> f16 [n][k] padded -----
__global__ void prep_kernel(const float* W1, const float* B1,
                            const float* W2, const float* B2,
                            const float* W3, const float* B3,
                            const float* W4, const float* B4,
                            char* wsblk, int in_dim) {
    _Float16* w1t = (_Float16*)(wsblk + OFF_W1T);
    _Float16* w2t = (_Float16*)(wsblk + OFF_W2T);
    _Float16* w3t = (_Float16*)(wsblk + OFF_W3T);
    float* b1 = (float*)(wsblk + OFF_B1);
    float* b2 = (float*)(wsblk + OFF_B2);
    float* b3 = (float*)(wsblk + OFF_B3);
    _Float16* w4h = (_Float16*)(wsblk + OFF_W4);
    float* b4 = (float*)(wsblk + OFF_B4);
    int t  = blockIdx.x * blockDim.x + threadIdx.x;
    int nt = gridDim.x * blockDim.x;
    for (int idx = t; idx < 128 * 32; idx += nt) {
        int n = idx >> 5, k = idx & 31;
        w1t[idx] = (k < in_dim) ? (_Float16)W1[k * 128 + n] : (_Float16)0.0f;
    }
    for (int idx = t; idx < 128 * 128; idx += nt) {
        int n = idx >> 7, k = idx & 127;
        w2t[idx] = (_Float16)W2[k * 128 + n];
        w3t[idx] = (_Float16)W3[k * 128 + n];
    }
    for (int idx = t; idx < 128; idx += nt) {
        b1[idx] = B1[idx]; b2[idx] = B2[idx]; b3[idx] = B3[idx];
        w4h[idx] = (_Float16)W4[idx];   // W4 shape (128,1)
    }
    if (t == 0) b4[0] = B4[0];
}

// ---------------- stage 1: message MLP over all (b,i,j) pairs ----------------
// grid = B * 7 blocks; block = (b, i-tile); wave w takes pairs (j0,j0+1),
// j0 = 2w + 16t -> covers j in [0,100) exactly once, both tiles always valid.
__global__ void __launch_bounds__(256)
msg_kernel(const float* obs, const char* wsmlp, float* msgbuf) {
    extern __shared__ char smem[];
    stage_params(wsmlp, smem);
    GET_PTRS();
    _Float16* hA = (_Float16*)(smem + OFF_HBUF) + wave * 4096;
    _Float16* hB = hA + 2048;
    int b = blockIdx.x / 7, itile = blockIdx.x % 7;
    int mrow = lane & 15;
    int ii  = itile * 16 + mrow;
    int iic = ii < 100 ? ii : 99;
    float obsi  = obs[b * 100 + iic];
    float coori = (float)ii * 0.01f;
    _Float16 z = (_Float16)0.0f;
    bool lo = lane < 16;     // lanes>=16 carry padded K=16..31 -> zero
    v16h abase;
#pragma unroll
    for (int r = 0; r < 16; ++r) abase[r] = z;
    abase[0] = lo ? (_Float16)obsi  : z;    // K0 = obs[b,i]
    abase[1] = lo ? (_Float16)coori : z;    // K1 = i/N
    float acc = 0.0f;
    for (int j0 = wave * 2; j0 < 100; j0 += 16) {
        float obsj0 = obs[b * 100 + j0];
        float obsj1 = obs[b * 100 + j0 + 1];
        v16h aA = abase, aB = abase;
        aA[2] = lo ? (_Float16)obsj0 : z;                       // K2 = obs[b,j]
        aA[3] = lo ? (_Float16)((float)j0 * 0.01f) : z;         // K3 = j/N
        aB[2] = lo ? (_Float16)obsj1 : z;
        aB[3] = lo ? (_Float16)((float)(j0 + 1) * 0.01f) : z;
        float2 r = mlp_tile2(aA, aB, w1t, w2t, w3t, tc, hA, hB, lane);
        acc += r.x + r.y;
    }
    if (lane < 16 && ii < 100) atomicAdd(&msgbuf[b * 100 + ii], acc);
}

// ---------------- stage 2a: actor MLP, tile rows = actions ------------------
__global__ void __launch_bounds__(256)
actor_kernel(const float* msgbuf, const char* wsmlp, float* logits) {
    extern __shared__ char smem[];
    stage_params(wsmlp, smem);
    GET_PTRS();
    _Float16* hbuf = (_Float16*)(smem + OFF_HBUF) + wave * 2048;
    int b = blockIdx.x;
    _Float16 z = (_Float16)0.0f;
    bool lo = lane < 16;
    v16h abase;
#pragma unroll
    for (int r = 0; r < 16; ++r) abase[r] = z;
    abase[2] = lo ? (_Float16)(float)(lane & 15) : z;   // K2 = action id
    float acc = 0.0f;
    for (int i = wave; i < 100; i += NWAVES) {
        float mv = msgbuf[b * 100 + i];
        float ci = (float)i * 0.01f;
        v16h a = abase;
        a[0] = lo ? (_Float16)mv : z;       // K0 = msg[b,i]
        a[1] = lo ? (_Float16)ci : z;       // K1 = i/N
        acc += mlp_tile(a, w1t, w2t, w3t, tc, hbuf, lane);
    }
    if (lane < 15) atomicAdd(&logits[b * 15 + lane], acc);
}

// ---------------- stage 2b: critic MLP, tile rows = nodes -------------------
__global__ void __launch_bounds__(256)
critic_kernel(const float* obs, const float* msgbuf, const char* wsmlp,
              float* value) {
    extern __shared__ char smem[];
    stage_params(wsmlp, smem);
    GET_PTRS();
    _Float16* hbuf = (_Float16*)(smem + OFF_HBUF) + wave * 2048;
    int b = blockIdx.x;
    if (wave < 7) {
        int mrow = lane & 15;
        int ii  = wave * 16 + mrow;
        int iic = ii < 100 ? ii : 99;
        float mv = msgbuf[b * 100 + iic];
        float ci = (float)ii * 0.01f;
        float ov = obs[b * 100 + iic];
        _Float16 z = (_Float16)0.0f;
        bool lo = lane < 16;
        v16h a;
#pragma unroll
        for (int r = 0; r < 16; ++r) a[r] = z;
        a[0] = lo ? (_Float16)mv : z;       // K0 = msg[b,i]
        a[1] = lo ? (_Float16)ci : z;       // K1 = i/N
        a[2] = lo ? (_Float16)ov : z;       // K2 = obs[b,i]
        float r = mlp_tile(a, w1t, w2t, w3t, tc, hbuf, lane);
        float c = (lane < 16 && ii < 100) ? r : 0.0f;
#pragma unroll
        for (int off = 16; off >= 1; off >>= 1) c += __shfl_xor(c, off, 32);
        if (lane == 0) atomicAdd(&value[b], c);
    }
}

// ---------------- host launch ----------------
extern "C" void kernel_launch(void* const* d_in, const int* in_sizes, int n_in,
                              void* d_out, int out_size, void* d_ws, size_t ws_size,
                              hipStream_t stream) {
    (void)in_sizes; (void)n_in; (void)ws_size;
    const float* obs = (const float*)d_in[0];
    char* ws = (char*)d_ws;
    float* msgbuf = (float*)(ws + WS_MSG);
    char* msgW  = ws + WS_PARAMS;
    char* actW  = ws + WS_PARAMS + MLP_STRIDE;
    char* critW = ws + WS_PARAMS + 2 * MLP_STRIDE;
    float* out = (float*)d_out;          // logits[32*15] then value[32]

    (void)hipMemsetAsync(d_out, 0, (size_t)out_size * sizeof(float), stream);
    (void)hipMemsetAsync(msgbuf, 0, 3200 * sizeof(float), stream);

    prep_kernel<<<8, 256, 0, stream>>>(
        (const float*)d_in[1], (const float*)d_in[2], (const float*)d_in[3],
        (const float*)d_in[4], (const float*)d_in[5], (const float*)d_in[6],
        (const float*)d_in[7], (const float*)d_in[8], msgW, 4);
    prep_kernel<<<8, 256, 0, stream>>>(
        (const float*)d_in[9], (const float*)d_in[10], (const float*)d_in[11],
        (const float*)d_in[12], (const float*)d_in[13], (const float*)d_in[14],
        (const float*)d_in[15], (const float*)d_in[16], actW, 3);
    prep_kernel<<<8, 256, 0, stream>>>(
        (const float*)d_in[17], (const float*)d_in[18], (const float*)d_in[19],
        (const float*)d_in[20], (const float*)d_in[21], (const float*)d_in[22],
        (const float*)d_in[23], (const float*)d_in[24], critW, 3);

    msg_kernel<<<32 * 7, 256, LDS_BYTES_2T, stream>>>(obs, msgW, msgbuf);
    actor_kernel<<<32, 256, LDS_BYTES_1T, stream>>>(msgbuf, actW, out);
    critic_kernel<<<32, 256, LDS_BYTES_1T, stream>>>(obs, msgbuf, critW, out + 480);
}
